// MultiHeadAttention_11038065951420
// MI455X (gfx1250) — compile-verified, matrix operand order checked by reading, not run on hardware
//
#include <hip/hip_runtime.h>
#include <hip/hip_bf16.h>

// ---------------- problem constants ----------------
constexpr int Bc = 4, Sc = 2048, Dc = 1024, Hc = 16, HDc = 64;
constexpr int Mc = Bc * Sc; // 8192 rows

// ---------------- vector types ----------------
typedef __attribute__((ext_vector_type(16))) __bf16 bf16x16;
typedef __attribute__((ext_vector_type(8)))  float  f32x8;
typedef __attribute__((ext_vector_type(4)))  float  f32x4;
typedef __attribute__((ext_vector_type(4)))  unsigned int u32x4;
typedef __attribute__((ext_vector_type(4)))  int i32x4;

union FragA { bf16x16 v; u32x4 u[2]; };

__device__ __forceinline__ unsigned short f2bf(float f) {
  unsigned int u = __builtin_bit_cast(unsigned int, f);
  u = (u + 0x7FFFu + ((u >> 16) & 1u)) >> 16; // round-to-nearest-even
  return (unsigned short)u;
}

// pack two f32 -> two bf16 in one dword (v_cvt_pk_bf16_f32 when available)
__device__ __forceinline__ unsigned pack_bf16(float a, float b) {
#if __has_builtin(__builtin_amdgcn_cvt_pk_bf16_f32)
  auto r = __builtin_amdgcn_cvt_pk_bf16_f32(a, b);
  return __builtin_bit_cast(unsigned, r);
#else
  return (unsigned)f2bf(a) | ((unsigned)f2bf(b) << 16);
#endif
}

// ---------------- async global->LDS staging (CDNA5 ASYNCcnt path) ----------
#if __has_builtin(__builtin_amdgcn_global_load_async_to_lds_b128) && \
    __has_builtin(__builtin_amdgcn_s_wait_asynccnt)
#define USE_ASYNC_LDS 1
#else
#define USE_ASYNC_LDS 0
#endif

__device__ __forceinline__ void stage16(const unsigned short* g,
                                        unsigned short* l) {
#if USE_ASYNC_LDS
  __builtin_amdgcn_global_load_async_to_lds_b128(
      (__attribute__((address_space(1))) i32x4*)g,
      (__attribute__((address_space(3))) i32x4*)l, 0, 0);
#else
  *(u32x4*)l = *(const u32x4*)g;
#endif
}
__device__ __forceinline__ void stage_wait() {
#if USE_ASYNC_LDS
  __builtin_amdgcn_s_wait_asynccnt(0);
#endif
}

// ---------------- DPP 16-lane butterfly reductions (pure VALU) ------------
template <int CTRL>
__device__ __forceinline__ float dppf(float x) {
  return __builtin_bit_cast(
      float, __builtin_amdgcn_update_dpp(0, __builtin_bit_cast(int, x), CTRL,
                                         0xF, 0xF, true));
}
__device__ __forceinline__ float redmax16(float x) {
  x = fmaxf(x, dppf<0xB1>(x));   // quad_perm [1,0,3,2]  (xor 1)
  x = fmaxf(x, dppf<0x4E>(x));   // quad_perm [2,3,0,1]  (xor 2)
  x = fmaxf(x, dppf<0x141>(x));  // row_half_mirror
  x = fmaxf(x, dppf<0x140>(x));  // row_mirror (completes 16-lane tree)
  return x;
}
__device__ __forceinline__ float redsum16(float x) {
  x += dppf<0xB1>(x);
  x += dppf<0x4E>(x);
  x += dppf<0x141>(x);
  x += dppf<0x140>(x);
  return x;
}

// =====================================================================
// Tiled GEMM:  C[M,N] = A[M,K] * W[K,N]   (K = N = Dc)
// MODE 0: A fp32 -> bf16*scale head-split [B,H,S,HD]  (Q gets scale=0.125)
// MODE 2: A fp32 -> bf16 head-split transposed [B,H,HD,S]   (V projection)
// MODE 1: A bf16 -> fp32 row-major [M,D]              (output projection)
// =====================================================================
template <int MODE>
__global__ __launch_bounds__(256) void gemm_kernel(const void* __restrict__ Ap,
                                                   const float* __restrict__ W,
                                                   void* __restrict__ Outp,
                                                   float scale) {
  constexpr int BM = 128, BN = 64, BK = 32;
  __shared__ __align__(16) unsigned short As[BM * BK]; // [BM][BK] row-major
  __shared__ __align__(16) unsigned short Bs[BN * BK]; // transposed: [BN][BK]

  const int tid  = threadIdx.x;
  const int lane = tid & 31;
  const int wid  = tid >> 5;
  const int wm   = wid & 3;
  const int wn   = wid >> 2;
  const int m0   = blockIdx.x * BM;
  const int n0   = blockIdx.y * BN;
  const int half = lane >> 4;
  const int l16  = lane & 15;

  f32x8 acc[2][2] = {};

  for (int k0 = 0; k0 < Dc; k0 += BK) {
    __syncthreads();
    { // ---- stage A tile ----
      const int row = tid >> 1;
      const int col = (tid & 1) * 16;
      if (MODE != 1) {
        const float* A = (const float*)Ap;
        const f32x4* src4 = (const f32x4*)(A + (size_t)(m0 + row) * Dc + k0 + col);
        u32x4 p0, p1;
#pragma unroll
        for (int jj = 0; jj < 2; ++jj) {
          f32x4 fv = src4[jj];
          p0[jj * 2]     = pack_bf16(fv[0], fv[1]);
          p0[jj * 2 + 1] = pack_bf16(fv[2], fv[3]);
          f32x4 gv = src4[2 + jj];
          p1[jj * 2]     = pack_bf16(gv[0], gv[1]);
          p1[jj * 2 + 1] = pack_bf16(gv[2], gv[3]);
        }
        *(u32x4*)&As[row * BK + col]     = p0;
        *(u32x4*)&As[row * BK + col + 8] = p1;
      } else {
        const unsigned short* A = (const unsigned short*)Ap;
        const unsigned short* src = A + (size_t)(m0 + row) * Dc + k0 + col;
        stage16(src,     &As[row * BK + col]);
        stage16(src + 8, &As[row * BK + col + 8]);
      }
    }
    { // ---- stage W tile (fp32 -> bf16, transposed into [BN][BK]) ----
      const int krow = tid >> 3;
      const int nc   = (tid & 7) * 8;
      const f32x4* wsrc = (const f32x4*)(W + (size_t)(k0 + krow) * Dc + n0 + nc);
      f32x4 w0 = wsrc[0], w1 = wsrc[1];
#pragma unroll
      for (int e = 0; e < 4; ++e) {
        Bs[(nc + e) * BK + krow]     = f2bf(w0[e]);
        Bs[(nc + 4 + e) * BK + krow] = f2bf(w1[e]);
      }
    }
    stage_wait();
    __syncthreads();

#pragma unroll
    for (int i = 0; i < 2; ++i) {
      const int arow = wm * 32 + i * 16 + l16;
      FragA a;
      a.u[0] = *(const u32x4*)&As[arow * BK + half * 8];
      a.u[1] = *(const u32x4*)&As[arow * BK + 16 + half * 8];
#pragma unroll
      for (int j = 0; j < 2; ++j) {
        const int brow = wn * 32 + j * 16 + l16;
        FragA bb;
        bb.u[0] = *(const u32x4*)&Bs[brow * BK + half * 8];
        bb.u[1] = *(const u32x4*)&Bs[brow * BK + 16 + half * 8];
        acc[i][j] = __builtin_amdgcn_wmma_f32_16x16x32_bf16(
            false, a.v, false, bb.v, (short)0, acc[i][j], false, false);
      }
    }
  }

  // ---- epilogue ----
  if (MODE == 2) {
    unsigned short* Out = (unsigned short*)Outp;
#pragma unroll
    for (int i = 0; i < 2; ++i)
#pragma unroll
      for (int j = 0; j < 2; ++j)
#pragma unroll
        for (int r = 0; r < 8; r += 2) {
          const int m = m0 + wm * 32 + i * 16 + r + 8 * half;
          const int n = n0 + wn * 32 + j * 16 + l16;
          const int b = m >> 11, s = m & (Sc - 1);
          const int h = n >> 6,  d = n & (HDc - 1);
          const unsigned pk = pack_bf16(acc[i][j][r], acc[i][j][r + 1]);
          *(unsigned*)&Out[(((size_t)b * Hc + h) * HDc + d) * Sc + s] = pk;
        }
  } else {
#pragma unroll
    for (int i = 0; i < 2; ++i)
#pragma unroll
      for (int j = 0; j < 2; ++j)
#pragma unroll
        for (int r = 0; r < 8; ++r) {
          const int m = m0 + wm * 32 + i * 16 + r + 8 * half;
          const int n = n0 + wn * 32 + j * 16 + l16;
          const float vv = acc[i][j][r];
          if (MODE == 0) {
            unsigned short* Out = (unsigned short*)Outp;
            const int b = m >> 11, s = m & (Sc - 1);
            const int h = n >> 6,  d = n & (HDc - 1);
            // scale folded here (0.125 for Q: exact exponent shift in bf16)
            Out[(((size_t)b * Hc + h) * Sc + s) * HDc + d] = f2bf(vv * scale);
          } else {
            float* Out = (float*)Outp;
            Out[(size_t)m * Dc + n] = vv;
          }
        }
  }
}

// =====================================================================
// Flash attention: block = 256 thr = 8 waves, all on one (b,h).
// Each wave owns TWO 16-row query tiles (32 rows); the block stages the
// 32-key K/V tiles via double-buffered async LDS copies. K/V fragments
// are reused across both query tiles -> 16 WMMAs per wave per key tile.
// Q is pre-scaled by 1/sqrt(hd) at projection time.
// =====================================================================
__global__ __launch_bounds__(256) void attn_kernel(
    const unsigned short* __restrict__ Qb, const unsigned short* __restrict__ Kb,
    const unsigned short* __restrict__ Vb, const int* __restrict__ vlenp,
    unsigned short* __restrict__ Ob) {
  __shared__ __align__(16) unsigned short Ks[2][32 * 64];    // [key][dim]
  __shared__ __align__(16) unsigned short Vs[2][64 * 32];    // [dim][key]
  __shared__ __align__(16) unsigned short Ps[8][2 * 16 * 32];// per-wave P

  const int tid  = threadIdx.x;
  const int lane = tid & 31;
  const int wid  = tid >> 5;

  const int bh   = blockIdx.x >> 3;       // 8 blocks per (b,h): S/256
  const int qblk = blockIdx.x & 7;
  const int row0 = qblk * 256 + wid * 32; // first of 32 query rows
  const int b    = bh >> 4;
  const int h    = bh & 15;
  const int vlen = vlenp[b];
  const int half = lane >> 4;
  const int l16  = lane & 15;

  const size_t headbase = (size_t)bh * Sc * HDc;
  const unsigned short* vbase = Vb + (size_t)bh * HDc * Sc; // [HD][S]

  // staging coordinates (8 contiguous elements per thread per tile)
  const int kkey = (tid * 8) >> 6, kdim = (tid * 8) & 63;   // K tile [32][64]
  const int vdim = (tid * 8) >> 5, vkey = (tid * 8) & 31;   // V tile [64][32]

  // ---- Q fragments: 2 query sub-tiles x 2 K-chunks, direct from global ----
  FragA qf[2][2];
#pragma unroll
  for (int g = 0; g < 2; ++g) {
    const unsigned short* qrow =
        Qb + headbase + (size_t)(row0 + g * 16 + l16) * HDc;
#pragma unroll
    for (int c = 0; c < 2; ++c) {
      qf[g][c].u[0] = *(const u32x4*)(qrow + c * 32 + half * 8);
      qf[g][c].u[1] = *(const u32x4*)(qrow + c * 32 + 16 + half * 8);
    }
  }

  float mrow[2][8], lrow[2][8];
  bool  rowmask[2][8];
#pragma unroll
  for (int g = 0; g < 2; ++g)
#pragma unroll
    for (int r = 0; r < 8; ++r) {
      mrow[g][r] = -1e30f; lrow[g][r] = 0.f;
      rowmask[g][r] = (row0 + g * 16 + r + 8 * half) >= vlen;
    }
  f32x8 oacc[2][4] = {};

  auto stage_tile = [&](int kt, int buf) {
    stage16(Kb + headbase + (size_t)(kt * 32 + kkey) * HDc + kdim,
            &Ks[buf][tid * 8]);
    stage16(vbase + (size_t)vdim * Sc + kt * 32 + vkey, &Vs[buf][tid * 8]);
  };

  constexpr int NT = Sc / 32;
  stage_tile(0, 0);

  for (int kt = 0; kt < NT; ++kt) {
    const int cur = kt & 1;
    stage_wait();        // this wave's async writes to buffer `cur` done
    __syncthreads();     // all waves' writes visible; prior reads retired
    if (kt + 1 < NT) stage_tile(kt + 1, cur ^ 1); // overlap with compute

    // ---- scores: 2 key sub-tiles x 2 chunks; kf reused for both g ----
    f32x8 sacc[2][2] = {};
#pragma unroll
    for (int j = 0; j < 2; ++j)
#pragma unroll
      for (int c = 0; c < 2; ++c) {
        FragA kf;
        const unsigned short* kr = Ks[cur] + (j * 16 + l16) * 64 + c * 32;
        kf.u[0] = *(const u32x4*)(kr + half * 8);
        kf.u[1] = *(const u32x4*)(kr + 16 + half * 8);
#pragma unroll
        for (int g = 0; g < 2; ++g)
          sacc[g][j] = __builtin_amdgcn_wmma_f32_16x16x32_bf16(
              false, qf[g][c].v, false, kf.v, (short)0, sacc[g][j], false,
              false);
      }

    // ---- online softmax (DPP-only row reductions; Q pre-scaled) ----
    float pm0[2][8], pm1[2][8];
#pragma unroll
    for (int g = 0; g < 2; ++g)
#pragma unroll
      for (int r = 0; r < 8; ++r) {
        const float s0 = rowmask[g][r] ? -100000.f : sacc[g][0][r];
        const float s1 = rowmask[g][r] ? -100000.f : sacc[g][1][r];
        const float mx   = redmax16(fmaxf(s0, s1));
        const float mnew = fmaxf(mrow[g][r], mx);
        const float corr = __expf(mrow[g][r] - mnew);
        const float p0 = __expf(s0 - mnew);
        const float p1 = __expf(s1 - mnew);
        const float rs = redsum16(p0 + p1);
        lrow[g][r] = lrow[g][r] * corr + rs;
        mrow[g][r] = mnew;
#pragma unroll
        for (int t = 0; t < 4; ++t) oacc[g][t][r] *= corr;
        pm0[g][r] = p0; pm1[g][r] = p1;
      }

    // ---- re-fragment P tiles through private LDS ----
    unsigned short* Pw = Ps[wid];
#pragma unroll
    for (int g = 0; g < 2; ++g)
#pragma unroll
      for (int r = 0; r < 8; ++r) {
        const int m = r + 8 * half;
        Pw[g * 512 + m * 32 + l16]      = f2bf(pm0[g][r]);
        Pw[g * 512 + m * 32 + 16 + l16] = f2bf(pm1[g][r]);
      }
    asm volatile("s_wait_dscnt 0" ::: "memory");

    FragA pf[2];
#pragma unroll
    for (int g = 0; g < 2; ++g) {
      pf[g].u[0] = *(const u32x4*)(Pw + g * 512 + l16 * 32 + half * 8);
      pf[g].u[1] = *(const u32x4*)(Pw + g * 512 + l16 * 32 + 16 + half * 8);
    }

    // ---- O += P * V ; vf reused for both g ----
#pragma unroll
    for (int t = 0; t < 4; ++t) {
      FragA vf;
      const unsigned short* vr = Vs[cur] + (t * 16 + l16) * 32;
      vf.u[0] = *(const u32x4*)(vr + half * 8);
      vf.u[1] = *(const u32x4*)(vr + 16 + half * 8);
#pragma unroll
      for (int g = 0; g < 2; ++g)
        oacc[g][t] = __builtin_amdgcn_wmma_f32_16x16x32_bf16(
            false, pf[g].v, false, vf.v, (short)0, oacc[g][t], false, false);
    }
  }

  // ---- normalize and store head-merged [B,S,H*HD] ----
#pragma unroll
  for (int g = 0; g < 2; ++g)
#pragma unroll
    for (int r = 0; r < 8; ++r) {
      const float inv = 1.f / lrow[g][r];
      const int srow = row0 + g * 16 + r + 8 * half;
#pragma unroll
      for (int t = 0; t < 4; ++t) {
        const int dim = t * 16 + l16;
        Ob[(((size_t)b * Sc + srow) * Hc + h) * HDc + dim] =
            f2bf(oacc[g][t][r] * inv);
      }
    }
}

// =====================================================================
extern "C" void kernel_launch(void* const* d_in, const int* in_sizes, int n_in,
                              void* d_out, int out_size, void* d_ws, size_t ws_size,
                              hipStream_t stream) {
  const float* q    = (const float*)d_in[0];
  const float* k    = (const float*)d_in[1];
  const float* v    = (const float*)d_in[2];
  const int*   vlen = (const int*)d_in[3];
  const float* Wq   = (const float*)d_in[4];
  const float* Wk   = (const float*)d_in[5];
  const float* Wv   = (const float*)d_in[6];
  const float* Wo   = (const float*)d_in[7];

  unsigned short* Qb = (unsigned short*)d_ws;          // [B,H,S,HD] bf16 (x0.125)
  unsigned short* Kb = Qb + (size_t)Mc * Dc;           // [B,H,S,HD] bf16
  unsigned short* Vb = Kb + (size_t)Mc * Dc;           // [B,H,HD,S] bf16
  unsigned short* Ob = Vb + (size_t)Mc * Dc;           // [B,S,H*HD] bf16

  dim3 ggrid(Mc / 128, Dc / 64);
  gemm_kernel<0><<<ggrid, 256, 0, stream>>>(q, Wq, Qb, 0.125f); // 1/sqrt(64)
  gemm_kernel<0><<<ggrid, 256, 0, stream>>>(k, Wk, Kb, 1.0f);
  gemm_kernel<2><<<ggrid, 256, 0, stream>>>(v, Wv, Vb, 1.0f);

  attn_kernel<<<Bc * Hc * (Sc / 256), 256, 0, stream>>>(Qb, Kb, Vb, vlen, Ob);

  gemm_kernel<1><<<ggrid, 256, 0, stream>>>(Ob, Wo, d_out, 1.0f);
}
